// MarketRegimeDetector_29317446762858
// MI455X (gfx1250) — matrix operand order repaired and checked by reading, not compiled, not run on hardware
//
#include <hip/hip_runtime.h>
#include <hip/hip_bf16.h>
#include <stdint.h>

// ---------------------------------------------------------------------------
// CDNA5 (gfx1250) wave32 WMMA bf16 implementation of MarketRegimeDetector.
// D=512, R=5, N tokens = in_sizes[0]/512.
// ---------------------------------------------------------------------------

typedef __attribute__((ext_vector_type(16))) __bf16    v16bf;
typedef __attribute__((ext_vector_type(8)))  float     v8f;
typedef __attribute__((ext_vector_type(8)))  uint32_t  v8u;

__device__ __forceinline__ uint16_t f2bf(float f) {
  uint32_t u = __float_as_uint(f);
  uint32_t r = u + 0x7FFFu + ((u >> 16) & 1u);   // round-to-nearest-even
  return (uint16_t)(r >> 16);
}
__device__ __forceinline__ uint32_t pack2bf(float lo, float hi) {
  return (uint32_t)f2bf(lo) | ((uint32_t)f2bf(hi) << 16);
}

__device__ __forceinline__ v8f wmma_bf16(const uint32_t a[8], const uint32_t b[8], v8f c) {
  v8u au, bu;
#pragma unroll
  for (int i = 0; i < 8; ++i) { au[i] = a[i]; bu[i] = b[i]; }
  v16bf av = __builtin_bit_cast(v16bf, au);
  v16bf bv = __builtin_bit_cast(v16bf, bu);
  // (neg_a, A, neg_b, B, c_mod, C, reuse_a, reuse_b)
  return __builtin_amdgcn_wmma_f32_16x16x32_bf16(false, av, false, bv, (short)0, c, false, false);
}

__device__ __forceinline__ void load_bfrag(const uint32_t* __restrict__ p, uint32_t b[8]) {
  uint4 lo = *(const uint4*)(p);
  uint4 hi = *(const uint4*)(p + 4);
  b[0] = lo.x; b[1] = lo.y; b[2] = lo.z; b[3] = lo.w;
  b[4] = hi.x; b[5] = hi.y; b[6] = hi.z; b[7] = hi.w;
}

// ---------------------------------------------------------------------------
// Pack fp32 weight [nmats, K, Nsrc] (row-major, [in,out]) into bf16 WMMA
// B-fragments. Fragment (mat, nt, kt) = 32 lanes x 8 dwords, lane-major.
// Within a fragment: element (k', n') -> lane = (((k'>>3)&1)<<4)|n',
// vgpr j = ((k'>>4)<<2)|((k'&7)>>1), half-slot = k'&1.
// ---------------------------------------------------------------------------
__global__ void mrd_pack_bfrags(const float* __restrict__ src, uint32_t* __restrict__ dst,
                                int K, int Nsrc, int Npad, int nmats) {
  long long e = (long long)blockIdx.x * blockDim.x + threadIdx.x;
  long long total = (long long)nmats * K * Npad;
  if (e >= total) return;
  int n   = (int)(e % Npad);
  int k   = (int)((e / Npad) % K);
  int mat = (int)(e / ((long long)Npad * K));
  float v = (n < Nsrc) ? src[(long long)mat * K * Nsrc + (long long)k * Nsrc + n] : 0.0f;
  int kt = k >> 5, nt = n >> 4;
  int kk = k & 31, nn = n & 15;
  int lane = (((kk >> 3) & 1) << 4) | nn;
  int j    = ((kk >> 4) << 2) | ((kk & 7) >> 1);
  int slot = kk & 1;
  int nkt  = K >> 5;
  long long fragbase = ((((long long)mat * (Npad >> 4) + nt) * nkt) + kt) * 256; // dwords
  uint16_t* p16 = (uint16_t*)dst;
  p16[(fragbase + (long long)lane * 8 + j) * 2 + slot] = f2bf(v);
}

// ---------------------------------------------------------------------------
// Detector: per-wave 16-token tile, 3 WMMA layers, softmax/argmax/confidence,
// bucket scatter. Block = 256 threads = 8 waves = 128 tokens.
// ---------------------------------------------------------------------------
__global__ __launch_bounds__(256)
void mrd_detector(const float* __restrict__ x,
                  const uint32_t* __restrict__ W1p, const float* __restrict__ b1,
                  const uint32_t* __restrict__ W2p, const float* __restrict__ b2,
                  const uint32_t* __restrict__ W3p, const float* __restrict__ b3,
                  const float* __restrict__ Wc, const float* __restrict__ bc,
                  float* __restrict__ probs_out, float* __restrict__ conf_out,
                  float* __restrict__ wgt, int* __restrict__ counts,
                  int* __restrict__ bucket, int Ntok) {
  __shared__ uint32_t smem[8 * 2048];                 // 8 waves x 8KB
  const int tid  = threadIdx.x;
  const int wv   = tid >> 5;
  const int lane = tid & 31;
  const int m    = lane & 15;                         // row (token-in-tile) / col
  const int g    = lane >> 4;
  uint32_t* wbuf = smem + wv * 2048;

  const int tokBase = blockIdx.x * 128 + wv * 16;
  const float* xrow = x + (size_t)(tokBase + m) * 512;

  // ---- A fragments of x: 16 ktiles (K=512), kept in VGPRs ----
  uint32_t a1[16][8];
#pragma unroll
  for (int kt = 0; kt < 16; ++kt) {
    int kb = kt * 32 + g * 8;
    float4 f0 = *(const float4*)(xrow + kb);
    float4 f1 = *(const float4*)(xrow + kb + 4);
    float4 f2 = *(const float4*)(xrow + kb + 16);
    float4 f3 = *(const float4*)(xrow + kb + 20);
    a1[kt][0] = pack2bf(f0.x, f0.y); a1[kt][1] = pack2bf(f0.z, f0.w);
    a1[kt][2] = pack2bf(f1.x, f1.y); a1[kt][3] = pack2bf(f1.z, f1.w);
    a1[kt][4] = pack2bf(f2.x, f2.y); a1[kt][5] = pack2bf(f2.z, f2.w);
    a1[kt][6] = pack2bf(f3.x, f3.y); a1[kt][7] = pack2bf(f3.z, f3.w);
  }

  // ---- Layer 1: [16x512]@[512x256], relu -> LDS h1 (bf16 [16][256]) ----
  for (int nt = 0; nt < 16; ++nt) {
    v8f acc = {};
    const uint32_t* bfr = W1p + (size_t)nt * 16 * 256 + lane * 8;
#pragma unroll
    for (int kt = 0; kt < 16; ++kt) {
      uint32_t b[8]; load_bfrag(bfr, b);
      acc = wmma_bf16(a1[kt], b, acc);
      bfr += 256;
    }
    float bias = b1[nt * 16 + m];
#pragma unroll
    for (int i = 0; i < 8; ++i) {
      float v = acc[i] + bias; v = v > 0.f ? v : 0.f;
      int mm = g * 8 + i;
      ((uint16_t*)wbuf)[mm * 256 + nt * 16 + m] = f2bf(v);
    }
  }
  __syncthreads();

  // ---- Layer 2: [16x256]@[256x128], relu -> LDS h2 (bf16 [16][128]) ----
  uint32_t a2[8][8];
#pragma unroll
  for (int kt = 0; kt < 8; ++kt) {
    int kb = kt * 32 + g * 8;
    const uint16_t* hrow = (const uint16_t*)wbuf + m * 256;
    uint4 lo = *(const uint4*)(hrow + kb);
    uint4 hi = *(const uint4*)(hrow + kb + 16);
    a2[kt][0] = lo.x; a2[kt][1] = lo.y; a2[kt][2] = lo.z; a2[kt][3] = lo.w;
    a2[kt][4] = hi.x; a2[kt][5] = hi.y; a2[kt][6] = hi.z; a2[kt][7] = hi.w;
  }
  __syncthreads();                                    // h1 fully consumed
  for (int nt = 0; nt < 8; ++nt) {
    v8f acc = {};
    const uint32_t* bfr = W2p + (size_t)nt * 8 * 256 + lane * 8;
#pragma unroll
    for (int kt = 0; kt < 8; ++kt) {
      uint32_t b[8]; load_bfrag(bfr, b);
      acc = wmma_bf16(a2[kt], b, acc);
      bfr += 256;
    }
    float bias = b2[nt * 16 + m];
#pragma unroll
    for (int i = 0; i < 8; ++i) {
      float v = acc[i] + bias; v = v > 0.f ? v : 0.f;
      ((uint16_t*)wbuf)[(g * 8 + i) * 128 + nt * 16 + m] = f2bf(v);
    }
  }
  __syncthreads();

  // ---- Layer 3: [16x128]@[128x16] (cols 0..4 valid) -> logits LDS f32 ----
  uint32_t a3[4][8];
#pragma unroll
  for (int kt = 0; kt < 4; ++kt) {
    int kb = kt * 32 + g * 8;
    const uint16_t* hrow = (const uint16_t*)wbuf + m * 128;
    uint4 lo = *(const uint4*)(hrow + kb);
    uint4 hi = *(const uint4*)(hrow + kb + 16);
    a3[kt][0] = lo.x; a3[kt][1] = lo.y; a3[kt][2] = lo.z; a3[kt][3] = lo.w;
    a3[kt][4] = hi.x; a3[kt][5] = hi.y; a3[kt][6] = hi.z; a3[kt][7] = hi.w;
  }
  {
    v8f acc = {};
    const uint32_t* bfr = W3p + lane * 8;
#pragma unroll
    for (int kt = 0; kt < 4; ++kt) {
      uint32_t b[8]; load_bfrag(bfr, b);
      acc = wmma_bf16(a3[kt], b, acc);
      bfr += 256;
    }
    float* lbuf = (float*)(wbuf + 1024);              // 16x16 f32
#pragma unroll
    for (int i = 0; i < 8; ++i) lbuf[(g * 8 + i) * 16 + m] = acc[i];
  }
  __syncthreads();

  // ---- Per-token epilogue (lanes 0..15 of each wave) ----
  if (lane < 16) {
    const float* lbuf = (const float*)(wbuf + 1024);
    int t = tokBase + lane;
    float lg[5];
#pragma unroll
    for (int j = 0; j < 5; ++j) lg[j] = lbuf[lane * 16 + j] + b3[j];
    float mx = lg[0];
#pragma unroll
    for (int j = 1; j < 5; ++j) mx = fmaxf(mx, lg[j]);
    float e[5], s = 0.f;
#pragma unroll
    for (int j = 0; j < 5; ++j) { e[j] = __expf(lg[j] - mx); s += e[j]; }
    float inv = 1.f / s;
    int best = 0; float bp = e[0];
#pragma unroll
    for (int j = 1; j < 5; ++j) if (e[j] > bp) { bp = e[j]; best = j; }  // first-max
    float z = bc[0];
#pragma unroll
    for (int j = 0; j < 5; ++j) {
      float pj = e[j] * inv;
      probs_out[(size_t)t * 5 + j] = pj;
      z += pj * Wc[j];
    }
    conf_out[t] = 1.f / (1.f + __expf(-z));
    wgt[t] = bp * inv;
    int pos = atomicAdd(&counts[best], 1);
    bucket[(size_t)best * Ntok + pos] = t;
  }
}

// ---------------------------------------------------------------------------
// Adapt: one block = 16 same-regime tokens x full D output.
// out = x + 0.2*p_top*(x @ Wr[r] + br[r]).  Grid (Ntok/16, 5), 128 threads.
// ---------------------------------------------------------------------------
__global__ __launch_bounds__(128)
void mrd_adapt(const float* __restrict__ x, const uint32_t* __restrict__ Wrp,
               const float* __restrict__ br, const float* __restrict__ wgt,
               const int* __restrict__ counts, const int* __restrict__ bucket,
               float* __restrict__ out, int Ntok) {
  __shared__ uint32_t xt[16 * 256];                   // 16 x 512 bf16 tile
  __shared__ int   toks[16];
  __shared__ float pw[16];

  const int r   = blockIdx.y;
  const int cnt = counts[r];
  const int s0  = blockIdx.x * 16;
  if (s0 >= cnt) return;                              // uniform across block

  const int tid = threadIdx.x;
  if (tid < 16) {
    int s = s0 + tid;
    int t = (s < cnt) ? bucket[(size_t)r * Ntok + s] : -1;
    toks[tid] = t;
    pw[tid]   = (t >= 0) ? 0.2f * wgt[t] : 0.0f;
  }
  __syncthreads();

  // Stage x tile (bf16) into LDS: 128 threads, 8 threads/row, 16 float4 each.
  {
    int row = tid >> 3, c8 = tid & 7;
    int t = toks[row];
    const float* xr = x + (size_t)(t < 0 ? 0 : t) * 512;
#pragma unroll
    for (int k = 0; k < 16; ++k) {
      int col = (c8 + k * 8) * 4;
      float4 f = make_float4(0.f, 0.f, 0.f, 0.f);
      if (t >= 0) f = *(const float4*)(xr + col);
      xt[row * 256 + (col >> 1)]     = pack2bf(f.x, f.y);
      xt[row * 256 + (col >> 1) + 1] = pack2bf(f.z, f.w);
    }
  }
  __syncthreads();

  const int lane = tid & 31, wv = tid >> 5;
  const int m = lane & 15, g = lane >> 4;

  // A fragments from LDS tile: 16 ktiles, kept in VGPRs across 8 ntiles.
  uint32_t a[16][8];
  const uint16_t* xrow16 = (const uint16_t*)xt + m * 512;
#pragma unroll
  for (int kt = 0; kt < 16; ++kt) {
    int kb = kt * 32 + g * 8;
    uint4 lo = *(const uint4*)(xrow16 + kb);
    uint4 hi = *(const uint4*)(xrow16 + kb + 16);
    a[kt][0] = lo.x; a[kt][1] = lo.y; a[kt][2] = lo.z; a[kt][3] = lo.w;
    a[kt][4] = hi.x; a[kt][5] = hi.y; a[kt][6] = hi.z; a[kt][7] = hi.w;
  }
  float prow[8]; int trow[8];
#pragma unroll
  for (int i = 0; i < 8; ++i) { prow[i] = pw[g * 8 + i]; trow[i] = toks[g * 8 + i]; }

  for (int nn = 0; nn < 8; ++nn) {
    int nt = wv * 8 + nn;
    v8f acc = {};
    const uint32_t* bfr = Wrp + ((size_t)r * 32 + nt) * 16 * 256 + lane * 8;
#pragma unroll
    for (int kt = 0; kt < 16; ++kt) {
      uint32_t b[8]; load_bfrag(bfr, b);
      acc = wmma_bf16(a[kt], b, acc);
      bfr += 256;
    }
    int col = nt * 16 + m;
    float bias = br[(size_t)r * 512 + col];
#pragma unroll
    for (int i = 0; i < 8; ++i) {
      int t = trow[i];
      if (t >= 0) {
        float xv = x[(size_t)t * 512 + col];
        out[(size_t)t * 512 + col] = xv + prow[i] * (acc[i] + bias);
      }
    }
  }
}

// ---------------------------------------------------------------------------
extern "C" void kernel_launch(void* const* d_in, const int* in_sizes, int n_in,
                              void* d_out, int out_size, void* d_ws, size_t ws_size,
                              hipStream_t stream) {
  (void)n_in; (void)out_size; (void)ws_size;
  const float* x  = (const float*)d_in[0];
  const float* W1 = (const float*)d_in[1];
  const float* b1 = (const float*)d_in[2];
  const float* W2 = (const float*)d_in[3];
  const float* b2 = (const float*)d_in[4];
  const float* W3 = (const float*)d_in[5];
  const float* b3 = (const float*)d_in[6];
  const float* Wr = (const float*)d_in[7];
  const float* br = (const float*)d_in[8];
  const float* Wc = (const float*)d_in[9];
  const float* bc = (const float*)d_in[10];

  const int Ntok = in_sizes[0] / 512;                 // 65536

  // workspace layout (dwords)
  uint32_t* W1p = (uint32_t*)d_ws;                    //  65536 dwords
  uint32_t* W2p = W1p + 65536;                        //  16384
  uint32_t* W3p = W2p + 16384;                        //   1024
  uint32_t* Wrp = W3p + 1024;                         // 655360
  float*    wgt = (float*)(Wrp + 655360);             // Ntok
  int*   counts = (int*)(wgt + Ntok);                 // 8
  int*   bucket = counts + 8;                         // 5*Ntok

  float* outx  = (float*)d_out;
  float* probs = outx + (size_t)Ntok * 512;
  float* conf  = probs + (size_t)Ntok * 5;

  hipMemsetAsync(counts, 0, 8 * sizeof(int), stream);

  mrd_pack_bfrags<<<(512 * 256 + 255) / 256, 256, 0, stream>>>(W1, W1p, 512, 256, 256, 1);
  mrd_pack_bfrags<<<(256 * 128 + 255) / 256, 256, 0, stream>>>(W2, W2p, 256, 128, 128, 1);
  mrd_pack_bfrags<<<(128 * 16 + 255) / 256, 256, 0, stream>>>(W3, W3p, 128, 5, 16, 1);
  mrd_pack_bfrags<<<(5 * 512 * 512 + 255) / 256, 256, 0, stream>>>(Wr, Wrp, 512, 512, 512, 5);

  mrd_detector<<<Ntok / 128, 256, 0, stream>>>(x, W1p, b1, W2p, b2, W3p, b3, Wc, bc,
                                               probs, conf, wgt, counts, bucket, Ntok);

  mrd_adapt<<<dim3(Ntok / 16, 5), 128, 0, stream>>>(x, Wrp, br, wgt, counts, bucket,
                                                    outx, Ntok);
}
